// MultiHeadAttention_58746562674823
// MI455X (gfx1250) — matrix operand order repaired and checked by reading, not compiled, not run on hardware
//
#include <hip/hip_runtime.h>
#include <cstdint>

// ---------------------------------------------------------------------------
// MHA for MI455X (gfx1250, wave32). bf16 WMMA + f32 accum, async-LDS staging
// for bf16-resident operands, double-buffered LDS tiles, NT stores for the
// huge attn tensor. Stages:
//   QKV proj  : fp32 GEMM -> bf16 heads (V stored transposed (B,H,D,T))
//   scores    : bf16 async GEMM, scale folded into Q, mask epilogue, NT store
//   softmax   : wave-per-row, NT store
//   ctx       : fp32(attn) x bf16-async(V^T) GEMM -> bf16 ctx (B,T,H*D)
//   out proj  : bf16-async(ctx) x fp32(Wo) GEMM + bias -> fp32 out
// ---------------------------------------------------------------------------

typedef __attribute__((ext_vector_type(16))) __bf16        v16bf;
typedef __attribute__((ext_vector_type(8)))  float         v8f;
typedef __attribute__((ext_vector_type(4)))  float         f32x4;
typedef __attribute__((ext_vector_type(4)))  unsigned int  u32x4;
typedef __attribute__((ext_vector_type(2)))  unsigned int  u32x2;
typedef int v4i_vs __attribute__((__vector_size__(4 * sizeof(int))));

constexpr int Bc = 8, Tc = 1024, DM = 1024, Hc = 16, Dc = 64;
constexpr int KS = 32;    // K depth per stage (one bf16 WMMA)
constexpr int LSTR = 40;  // LDS row stride in bf16 elems (80 B, conflict-free)

#if __has_builtin(__builtin_amdgcn_global_load_async_to_lds_b128)
#define HAVE_ASYNC 1
#else
#define HAVE_ASYNC 0
#endif

__device__ __forceinline__ unsigned short f2bf(float x) {
  unsigned u = __float_as_uint(x);
  return (unsigned short)((u + 0x7FFFu + ((u >> 16) & 1u)) >> 16);  // RNE
}
__device__ __forceinline__ unsigned pk2(float a, float b) {
  return (unsigned)f2bf(a) | ((unsigned)f2bf(b) << 16);
}

// 16B global -> LDS copy: async (ASYNCcnt) when available, else via VGPRs.
__device__ __forceinline__ void copy16_async(const unsigned short* g,
                                             unsigned short* l) {
#if HAVE_ASYNC
  __builtin_amdgcn_global_load_async_to_lds_b128(
      (__attribute__((address_space(1))) v4i_vs*)(uintptr_t)g,
      (__attribute__((address_space(3))) v4i_vs*)(uintptr_t)l, 0, 0);
#else
  *(u32x4*)l = *(const u32x4*)g;
#endif
}

__device__ __forceinline__ void wait_async0() {
#if HAVE_ASYNC
#if __has_builtin(__builtin_amdgcn_s_wait_asynccnt)
  __builtin_amdgcn_s_wait_asynccnt(0);
#else
  asm volatile("s_wait_asynccnt 0x0" ::: "memory");
#endif
#endif
}

union FragBF { v16bf v; u32x4 q[2]; };

enum class AMode { F32, BF16 };
enum class BMode { NT_F32, NT_BF16 };

// ---- tile stagers (ROWS x 32 elems into LDS, row stride LSTR) --------------
template <int ROWS>
__device__ __forceinline__ void stage_f32(unsigned short* Ls, const float* G,
                                          long r0, int ld, int kb, int tid) {
  constexpr int L = ROWS * 8 / 256;  // float4 loads per thread
#pragma unroll
  for (int i = 0; i < L; i++) {
    int idx = tid + i * 256, r = idx >> 3, c4 = idx & 7;
    f32x4 f = *(const f32x4*)(G + (r0 + r) * (long)ld + kb + 4 * c4);
    u32x2 p;
    p.x = pk2(f.x, f.y);
    p.y = pk2(f.z, f.w);
    *(u32x2*)&Ls[r * LSTR + 4 * c4] = p;
  }
}

template <int ROWS>
__device__ __forceinline__ void stage_bf16(unsigned short* Ls,
                                           const unsigned short* G, long r0,
                                           int ld, int kb, int tid) {
  constexpr int L = ROWS * 4 / 256;  // 16B chunks per thread
#pragma unroll
  for (int i = 0; i < L; i++) {
    int idx = tid + i * 256, r = idx >> 2, c = idx & 3;
    copy16_async(G + (r0 + r) * (long)ld + kb + 8 * c, &Ls[r * LSTR + 8 * c]);
  }
}

// ---------------------------------------------------------------------------
// Generic tiled GEMM: C[m,n] = sum_k A[m,k] * B[n,k]   (B is NT: (N,K))
// Double-buffered LDS; async staging for bf16 operands. Exact tiling assumed.
// ---------------------------------------------------------------------------
template <int BM, int BN, int WAVES_M, int WAVES_N, AMode AM, BMode BMD,
          class Epi>
__device__ __forceinline__ void gemm_core(const void* Ap, int lda,
                                          const void* Bp, int ldb, int K,
                                          Epi epi) {
  constexpr int WM = BM / WAVES_M, WN = BN / WAVES_N;
  constexpr int WMT = WM / 16, WNT = WN / 16;
  static_assert(WAVES_M * WAVES_N == 8, "256 threads = 8 waves");

  __shared__ unsigned short As[2 * BM * LSTR];
  __shared__ unsigned short Bs[2 * BN * LSTR];

  const int tid = threadIdx.x;
  const int lane = tid & 31, wave = tid >> 5;
  const int wm = wave / WAVES_N, wn = wave % WAVES_N;
  const int l16 = lane & 15, lh = lane >> 4;
  const long m0 = (long)blockIdx.y * BM, n0 = (long)blockIdx.x * BN;

  auto stageA = [&](int buf, int kb) {
    if constexpr (AM == AMode::F32)
      stage_f32<BM>(&As[buf * BM * LSTR], (const float*)Ap, m0, lda, kb, tid);
    else
      stage_bf16<BM>(&As[buf * BM * LSTR], (const unsigned short*)Ap, m0, lda,
                     kb, tid);
  };
  auto stageB = [&](int buf, int kb) {
    if constexpr (BMD == BMode::NT_F32)
      stage_f32<BN>(&Bs[buf * BN * LSTR], (const float*)Bp, n0, ldb, kb, tid);
    else
      stage_bf16<BN>(&Bs[buf * BN * LSTR], (const unsigned short*)Bp, n0, ldb,
                     kb, tid);
  };

  v8f acc[WMT][WNT];
  v8f zero = {};
#pragma unroll
  for (int i = 0; i < WMT; i++)
#pragma unroll
    for (int j = 0; j < WNT; j++) acc[i][j] = zero;

  const int NIT = K / KS;
  stageA(0, 0);
  stageB(0, 0);
  wait_async0();
  __syncthreads();

  for (int it = 0; it < NIT; ++it) {
    const int cur = it & 1, nxt = cur ^ 1;
    if (it + 1 < NIT) {  // overlap next tile's copies with this tile's math
      stageA(nxt, (it + 1) * KS);
      stageB(nxt, (it + 1) * KS);
    }
    const unsigned short* Ab = &As[cur * BM * LSTR];
    const unsigned short* Bb = &Bs[cur * BN * LSTR];

    FragBF a[WMT], b[WNT];
#pragma unroll
    for (int i = 0; i < WMT; i++) {
      // A 16x32 bf16 (ISA 7.12.2): lane half selects K base 0/8; e>=8 adds 16
      const unsigned short* p = &Ab[(wm * WM + i * 16 + l16) * LSTR];
      a[i].q[0] = *(const u32x4*)(p + 8 * lh);
      a[i].q[1] = *(const u32x4*)(p + 16 + 8 * lh);
    }
#pragma unroll
    for (int j = 0; j < WNT; j++) {
      // B 32x16 bf16: same K order as A: lane half selects K base 0/8; e>=8 adds 16
      const unsigned short* p = &Bb[(wn * WN + j * 16 + l16) * LSTR + 8 * lh];
      b[j].q[0] = *(const u32x4*)(p);
      b[j].q[1] = *(const u32x4*)(p + 16);
    }
#pragma unroll
    for (int i = 0; i < WMT; i++)
#pragma unroll
      for (int j = 0; j < WNT; j++)
        acc[i][j] = __builtin_amdgcn_wmma_f32_16x16x32_bf16(
            false, a[i].v, false, b[j].v, (short)0, acc[i][j], false, false);

    if (it + 1 < NIT) wait_async0();
    __syncthreads();
  }

  // C/D layout: M = r + 8*(lane/16), N = lane%16
#pragma unroll
  for (int i = 0; i < WMT; i++)
#pragma unroll
    for (int j = 0; j < WNT; j++)
#pragma unroll
      for (int r = 0; r < 8; r++)
        epi((int)(m0 + wm * WM + i * 16 + lh * 8 + r),
            (int)(n0 + wn * WN + j * 16 + l16), acc[i][j][r]);
}

// ---------------------------------------------------------------------------
// Epilogues
// ---------------------------------------------------------------------------
struct EpiHeadsBF16 {  // y = (x+bias)*alpha -> bf16 (B,H,T,D) or (B,H,D,T)
  unsigned short* out;
  const float* bias;
  float alpha;
  int trans;  // 1: V^T layout (B,H,D,T)
  __device__ void operator()(int m, int n, float v) const {
    v = (v + bias[n]) * alpha;
    int b = m >> 10, t = m & 1023, h = n >> 6, d = n & 63;
    long idx = trans ? (((long)(b * Hc + h) * Dc + d) << 10) + t
                     : (((long)(b * Hc + h) * Tc + t) << 6) + d;
    out[idx] = f2bf(v);
  }
};

struct EpiScores {  // masked scores -> fp32 attn (NT store), ldc = T
  float* C;
  const int* mask;
  __device__ void operator()(int m, int n, float v) const {
    if (mask[(long)m * Tc + n] == 0) v = -1.0e9f;
    __builtin_nontemporal_store(v, &C[(long)m * Tc + n]);
  }
};

struct EpiCtxBF16 {  // ctx -> bf16 (B,T,H*D)
  unsigned short* C;
  int ldc;
  __device__ void operator()(int m, int n, float v) const {
    C[(long)m * ldc + n] = f2bf(v);
  }
};

struct EpiOut {  // out = x + bias -> fp32
  float* C;
  int ldc;
  const float* bias;
  __device__ void operator()(int m, int n, float v) const {
    C[(long)m * ldc + n] = v + bias[n];
  }
};

// ---------------------------------------------------------------------------
// Kernels
// ---------------------------------------------------------------------------
__global__ __launch_bounds__(256) void k_proj(const float* __restrict__ X,
                                              const float* __restrict__ W,
                                              const float* __restrict__ bias,
                                              unsigned short* __restrict__ dst,
                                              float alpha, int trans) {
  gemm_core<128, 128, 2, 4, AMode::F32, BMode::NT_F32>(
      X, DM, W, DM, DM, EpiHeadsBF16{dst, bias, alpha, trans});
}

__global__ __launch_bounds__(256) void k_scores(
    const unsigned short* __restrict__ qh, const unsigned short* __restrict__ kh,
    float* __restrict__ attn, const int* __restrict__ mask) {
  long z = blockIdx.z;  // b*H + h
  gemm_core<128, 128, 2, 4, AMode::BF16, BMode::NT_BF16>(
      qh + z * Tc * Dc, Dc, kh + z * Tc * Dc, Dc, Dc,
      EpiScores{attn + z * Tc * Tc, mask});
}

__global__ __launch_bounds__(256) void k_softmax(float* __restrict__ attn) {
  const long row = (long)blockIdx.x * 8 + (threadIdx.x >> 5);
  const int lane = threadIdx.x & 31;
  float* p = attn + row * Tc;
  f32x4 v[8];
  float mx = -3.4e38f;
#pragma unroll
  for (int i = 0; i < 8; i++) {
    v[i] = *(const f32x4*)(p + (lane + 32 * i) * 4);
    mx = fmaxf(mx, fmaxf(fmaxf(v[i].x, v[i].y), fmaxf(v[i].z, v[i].w)));
  }
#pragma unroll
  for (int o = 16; o > 0; o >>= 1) mx = fmaxf(mx, __shfl_xor(mx, o, 32));
  float s = 0.f;
#pragma unroll
  for (int i = 0; i < 8; i++) {
    v[i].x = __expf(v[i].x - mx);
    v[i].y = __expf(v[i].y - mx);
    v[i].z = __expf(v[i].z - mx);
    v[i].w = __expf(v[i].w - mx);
    s += v[i].x + v[i].y + v[i].z + v[i].w;
  }
#pragma unroll
  for (int o = 16; o > 0; o >>= 1) s += __shfl_xor(s, o, 32);
  const float inv = 1.0f / s;
#pragma unroll
  for (int i = 0; i < 8; i++) {
    v[i] *= inv;
    __builtin_nontemporal_store(v[i], (f32x4*)(p + (lane + 32 * i) * 4));
  }
}

__global__ __launch_bounds__(256) void k_ctx(
    const float* __restrict__ attn, const unsigned short* __restrict__ vhT,
    unsigned short* __restrict__ ctx) {
  long z = blockIdx.z;
  int b = (int)(z / Hc), h = (int)(z % Hc);
  // A = attn (T,T) fp32; B = V^T (D,T) bf16 async; C -> bf16 (B,T,H*D)
  gemm_core<128, 64, 4, 2, AMode::F32, BMode::NT_BF16>(
      attn + z * Tc * Tc, Tc, vhT + z * Dc * Tc, Tc, Tc,
      EpiCtxBF16{ctx + (long)b * Tc * (Hc * Dc) + h * Dc, Hc * Dc});
}

__global__ __launch_bounds__(256) void k_out(
    const unsigned short* __restrict__ ctx, const float* __restrict__ Wo,
    const float* __restrict__ bo, float* __restrict__ out) {
  gemm_core<128, 64, 4, 2, AMode::BF16, BMode::NT_F32>(
      ctx, Hc * Dc, Wo, Hc * Dc, Hc * Dc, EpiOut{out, Dc, bo});
}

// ---------------------------------------------------------------------------
// Host launcher
// ---------------------------------------------------------------------------
extern "C" void kernel_launch(void* const* d_in, const int* in_sizes, int n_in,
                              void* d_out, int out_size, void* d_ws,
                              size_t ws_size, hipStream_t stream) {
  const float* q  = (const float*)d_in[0];
  const float* k  = (const float*)d_in[1];
  const float* v  = (const float*)d_in[2];
  const float* Wq = (const float*)d_in[3];
  const float* bq = (const float*)d_in[4];
  const float* Wk = (const float*)d_in[5];
  const float* bk = (const float*)d_in[6];
  const float* Wv = (const float*)d_in[7];
  const float* bv = (const float*)d_in[8];
  const float* Wo = (const float*)d_in[9];
  const float* bo = (const float*)d_in[10];
  const int* mask = (const int*)d_in[11];

  float* out  = (float*)d_out;
  float* attn = out + (long)Bc * Tc * Dc;  // outputs: [out | attn]

  unsigned short* wsu = (unsigned short*)d_ws;
  const long NQ = (long)Bc * Hc * Tc * Dc;  // 8,388,608 elems
  unsigned short* qh  = wsu;                // bf16 (B,H,T,D)
  unsigned short* kh  = wsu + NQ;           // bf16 (B,H,T,D)
  unsigned short* vhT = wsu + 2 * NQ;       // bf16 (B,H,D,T)
  unsigned short* ctx = wsu;                // bf16 (B,T,H*D), reuses qh

  dim3 blk(256);
  const float scale = 0.125f;  // 1/sqrt(D), folded into Q projection

  k_proj<<<dim3(DM / 128, (Bc * Tc) / 128, 1), blk, 0, stream>>>(q, Wq, bq, qh,
                                                                 scale, 0);
  k_proj<<<dim3(DM / 128, (Bc * Tc) / 128, 1), blk, 0, stream>>>(k, Wk, bk, kh,
                                                                 1.0f, 0);
  k_proj<<<dim3(DM / 128, (Bc * Tc) / 128, 1), blk, 0, stream>>>(v, Wv, bv, vhT,
                                                                 1.0f, 1);
  k_scores<<<dim3(Tc / 128, Tc / 128, Bc * Hc), blk, 0, stream>>>(qh, kh, attn,
                                                                  mask);
  k_softmax<<<dim3((Bc * Hc * Tc) / 8), blk, 0, stream>>>(attn);
  k_ctx<<<dim3(1, Tc / 128, Bc * Hc), blk, 0, stream>>>(attn, vhT, ctx);
  k_out<<<dim3(1, (Bc * Tc) / 128, 1), blk, 0, stream>>>(ctx, Wo, bo, out);

  (void)in_sizes; (void)n_in; (void)out_size; (void)ws_size;
}
